// GAT_48859547959356
// MI455X (gfx1250) — compile-verified, matrix operand order checked by reading, not run on hardware
//
#include <hip/hip_runtime.h>
#include <math.h>

// Model dims (from reference)
#define S_ 100
#define D_ 20
#define T_ 30
#define H_ 64
#define G_ 192      // 3*H gates
#define FT_ 512
#define NH_ 8
#define ALPHA_ 0.2f

typedef __attribute__((ext_vector_type(16))) _Float16 v16h;
typedef __attribute__((ext_vector_type(8)))  _Float16 v8h;
typedef __attribute__((ext_vector_type(4)))  _Float16 v4h;
typedef __attribute__((ext_vector_type(8)))  float    v8f;

__device__ __forceinline__ float sigmoidf_(float x) { return 1.0f / (1.0f + expf(-x)); }

// ---- CDNA5 async global->LDS copies (CDNA5 ISA) ----
// LDS byte offset == low 32 bits of the generic shared pointer.
__device__ __forceinline__ void async_ld_b128(void* lds, const void* g) {
  unsigned off = (unsigned)(size_t)lds;
  asm volatile("global_load_async_to_lds_b128 %0, %1, off" ::"v"(off), "v"(g) : "memory");
}
__device__ __forceinline__ void async_ld_b32(void* lds, const void* g) {
  unsigned off = (unsigned)(size_t)lds;
  asm volatile("global_load_async_to_lds_b32 %0, %1, off" ::"v"(off), "v"(g) : "memory");
}
__device__ __forceinline__ void wait_async0() {
  asm volatile("s_wait_asynccnt 0x0" ::: "memory");
}

// ---- WMMA 16x16x32 f16 fragment helpers (CDNA5 wave32 layouts, CDNA5 ISA) ----
// Lane striping: m/n = lane&15, kb = (lane>>4)*8; VGPR r<4 -> K pair kb+2r; r>=4 -> kb+16+2(r-4).
// Each lane needs halves [kb..kb+7] and [kb+16..kb+23] of its row: two 16B ds_load_b128.

// A: 16x32 (MxK), row-major [M][K] in LDS. lda elements; 16B-aligned rows required.
__device__ __forceinline__ v16h load_frag_A(const _Float16* p, int lda, int row0, int k0,
                                            int lane) {
  const _Float16* q = p + (row0 + (lane & 15)) * lda + k0 + ((lane >> 4) << 3);
  v8h lo = *(const v8h*)q;
  v8h hi = *(const v8h*)(q + 16);
  return __builtin_shufflevector(lo, hi, 0, 1, 2, 3, 4, 5, 6, 7, 8, 9, 10, 11, 12, 13, 14, 15);
}
// B: 32x16 (KxN) stored TRANSPOSED in LDS as Bt[n][k] (ldk elements per row).
// For gate GEMMs (x @ W^T) this is just W's native row-major layout.
__device__ __forceinline__ v16h load_frag_Bt(const _Float16* p, int ldk, int k0, int col0,
                                             int lane) {
  const _Float16* q = p + (col0 + (lane & 15)) * ldk + k0 + ((lane >> 4) << 3);
  v8h lo = *(const v8h*)q;
  v8h hi = *(const v8h*)(q + 16);
  return __builtin_shufflevector(lo, hi, 0, 1, 2, 3, 4, 5, 6, 7, 8, 9, 10, 11, 12, 13, 14, 15);
}
// C/D: lane n = lane&15, row m = (lane>>4)*8 + r for VGPR r.

__device__ __forceinline__ v4h cvt4(float4 v) {
  v4h h;
  h[0] = (_Float16)v.x; h[1] = (_Float16)v.y; h[2] = (_Float16)v.z; h[3] = (_Float16)v.w;
  return h;
}

// =====================================================================================
// Kernel A: text GRU input projection. One block per (stock, day).
// gi[(s,d), t, g] = text[s,d,t,:] . Wih[s,g,:] + bih[s,g]   -> [30,192] via WMMA
// M=32(pad of 30), N=192, K=512 chunked by 64. Wt holds Wih rows (== B transposed).
// text_input is the 123 MB stream that bounds this phase -> prefetch next K-chunk.
// =====================================================================================
__global__ void __launch_bounds__(256) text_gi_kernel(const float* __restrict__ X,
                                                      const float* __restrict__ Wih,
                                                      const float* __restrict__ bih,
                                                      float* __restrict__ gi_out) {
  __shared__ __align__(16) _Float16 Xh[32 * 64];    // 4 KB, A tile
  __shared__ __align__(16) _Float16 Wt[G_ * 64];    // 24.5 KB, Wt[n][k] = Wih[n][kglobal]
  __shared__ __align__(16) float    stage[32 * G_]; // 24.5 KB output staging
  int blk = blockIdx.x;                // s*20 + d
  int s = blk / D_;
  int tid = threadIdx.x, lane = tid & 31, wid = tid >> 5;
  const float* Xp = X + (size_t)blk * T_ * FT_;
  const float* Wp = Wih + (size_t)s * G_ * FT_;

  v8f acc[3] = {};                     // 24 output tiles / 8 waves = 3 per wave
  for (int kc = 0; kc < FT_ / 64; ++kc) {
    for (int c = tid; c < 32 * 16; c += 256) {          // float4-vectorized A tile load
      int m = c >> 4, f = (c & 15) << 2;
      float4 v = make_float4(0.f, 0.f, 0.f, 0.f);
      if (m < T_) {
        v = *(const float4*)(Xp + m * FT_ + kc * 64 + f);
        if (kc < 7) __builtin_prefetch(Xp + m * FT_ + (kc + 1) * 64 + f, 0, 0);
      }
      *(v4h*)(Xh + m * 64 + f) = cvt4(v);
    }
    for (int c = tid; c < G_ * 16; c += 256) {          // coalesced W rows -> Bt layout
      int n = c >> 4, f = (c & 15) << 2;
      float4 v = *(const float4*)(Wp + n * FT_ + kc * 64 + f);
      *(v4h*)(Wt + n * 64 + f) = cvt4(v);
    }
    __syncthreads();
#pragma unroll
    for (int i = 0; i < 3; ++i) {
      int tile = wid * 3 + i;
      int mt = tile / 12, nt = tile % 12;
#pragma unroll
      for (int ks = 0; ks < 2; ++ks) {
        v16h a = load_frag_A(Xh, 64, mt * 16, ks * 32, lane);
        v16h b = load_frag_Bt(Wt, 64, ks * 32, nt * 16, lane);
        acc[i] = __builtin_amdgcn_wmma_f32_16x16x32_f16(false, a, false, b, (short)0, acc[i],
                                                        false, false);
      }
    }
    __syncthreads();
  }
  // stage D tiles in LDS, then coalesced float4 global store (+bih)
  int nl = lane & 15, mb = (lane >> 4) << 3;
#pragma unroll
  for (int i = 0; i < 3; ++i) {
    int tile = wid * 3 + i;
    int mt = tile / 12, nt = tile % 12;
#pragma unroll
    for (int r = 0; r < 8; ++r) stage[(mt * 16 + mb + r) * G_ + nt * 16 + nl] = acc[i][r];
  }
  __syncthreads();
  const float* bp = bih + s * G_;
  for (int c = tid; c < T_ * (G_ / 4); c += 256) {  // 30 * 48 float4 chunks
    int m = c / 48, f = (c % 48) << 2;
    float4 v = *(const float4*)(stage + m * G_ + f);
    float4 b = *(const float4*)(bp + f);
    v.x += b.x; v.y += b.y; v.z += b.z; v.w += b.w;
    *(float4*)(gi_out + ((size_t)blk * T_ + m) * G_ + f) = v;
  }
}

// =====================================================================================
// Kernel B: text GRU recurrence. One block per stock; the 20 day-sequences are
// independent, so each timestep is a batched GEMM: GH = H[20,64] @ Whh^T[64,192] (WMMA).
// Whr = Whh rows in f16 (== B transposed-in-LDS layout).
// =====================================================================================
__global__ void __launch_bounds__(256) text_rec_kernel(const float* __restrict__ gi,
                                                       const float* __restrict__ Whh,
                                                       const float* __restrict__ bhh,
                                                       float* __restrict__ outs,
                                                       float* __restrict__ hlast) {
  __shared__ __align__(16) _Float16 Whr[G_ * 64];  // 24.5 KB  Whr[g][k] = Whh[g][k]
  __shared__ __align__(16) _Float16 Hh[32 * 64];   // 4 KB     f16 state (A frags), rows>=20 zero
  __shared__ float Hf[20 * 64];                    // 5 KB     f32 state
  __shared__ float GH[32 * G_];                    // 24.5 KB  gate pre-activations
  int s = blockIdx.x;
  int tid = threadIdx.x, lane = tid & 31, wid = tid >> 5;

  const float* Wp = Whh + (size_t)s * G_ * H_;
  for (int c = tid; c < G_ * 16; c += 256) {   // coalesced float4 -> packed f16
    int f = c << 2;
    *(v4h*)(Whr + f) = cvt4(*(const float4*)(Wp + f));
  }
  for (int idx = tid; idx < 32 * 64; idx += 256) Hh[idx] = (_Float16)0.0f;
  for (int idx = tid; idx < 20 * 64; idx += 256) Hf[idx] = 0.0f;
  __syncthreads();

  for (int t = 0; t < T_; ++t) {
#pragma unroll
    for (int i = 0; i < 3; ++i) {           // 24 tiles / 8 waves
      int tile = wid * 3 + i;
      int mt = tile / 12, nt = tile % 12;
      v8f acc = {};
#pragma unroll
      for (int ks = 0; ks < 2; ++ks) {
        v16h a = load_frag_A(Hh, 64, mt * 16, ks * 32, lane);
        v16h b = load_frag_Bt(Whr, 64, ks * 32, nt * 16, lane);
        acc = __builtin_amdgcn_wmma_f32_16x16x32_f16(false, a, false, b, (short)0, acc, false,
                                                     false);
      }
      int nl = lane & 15, mb = (lane >> 4) << 3;
#pragma unroll
      for (int r = 0; r < 8; ++r) GH[(mt * 16 + mb + r) * G_ + nt * 16 + nl] = acc[r];
    }
    __syncthreads();
    for (int idx = tid; idx < D_ * H_; idx += 256) {
      int d = idx >> 6, j = idx & 63;
      size_t base = ((size_t)(s * D_ + d) * T_ + t) * G_;
      float ir = gi[base + j], iz = gi[base + 64 + j], in_ = gi[base + 128 + j];
      float hr = GH[d * G_ + j] + bhh[s * G_ + j];
      float hz = GH[d * G_ + 64 + j] + bhh[s * G_ + 64 + j];
      float hn = GH[d * G_ + 128 + j] + bhh[s * G_ + 128 + j];
      float r = sigmoidf_(ir + hr);
      float z = sigmoidf_(iz + hz);
      float n = tanhf(in_ + r * hn);
      float h = (1.0f - z) * n + z * Hf[idx];
      Hf[idx] = h;
      Hh[idx] = (_Float16)h;
      outs[((size_t)(s * D_ + d) * T_ + t) * H_ + j] = h;
    }
    __syncthreads();
  }
  for (int idx = tid; idx < D_ * H_; idx += 256) hlast[(size_t)s * D_ * H_ + idx] = Hf[idx];
}

// =====================================================================================
// Kernel C: text attention pool per (stock, day): news = softmax(tanh(O@Wa).h_last) @ O
// Staging copies use the CDNA5 async global->LDS path (no VGPR round-trip).
// =====================================================================================
__global__ void __launch_bounds__(128) attn_text_kernel(const float* __restrict__ outs,
                                                        const float* __restrict__ hlast,
                                                        const float* __restrict__ taW,
                                                        float* __restrict__ news) {
  __shared__ __align__(16) float O[T_ * H_];
  __shared__ __align__(16) float Wa[H_ * H_];
  __shared__ __align__(16) float hl[H_];
  __shared__ float P[T_ * H_];
  __shared__ float sc[T_];
  __shared__ float aw[T_];
  int blk = blockIdx.x, s = blk / D_;
  int tid = threadIdx.x;
  const float* Op = outs + (size_t)blk * T_ * H_;
  const float* Wp = taW + (size_t)s * H_ * H_;
  for (int c = tid; c < (T_ * H_) / 4; c += 128) async_ld_b128(O + c * 4, Op + c * 4);
  for (int c = tid; c < (H_ * H_) / 4; c += 128) async_ld_b128(Wa + c * 4, Wp + c * 4);
  if (tid < H_ / 4) async_ld_b128(hl + tid * 4, hlast + (size_t)blk * H_ + tid * 4);
  wait_async0();
  __syncthreads();
  for (int idx = tid; idx < T_ * H_; idx += 128) {
    int t = idx >> 6, j = idx & 63;
    float a = 0.0f;
    for (int k = 0; k < H_; ++k) a += O[t * H_ + k] * Wa[k * H_ + j];
    P[idx] = tanhf(a);
  }
  __syncthreads();
  if (tid < T_) {
    float a = 0.0f;
    for (int j = 0; j < H_; ++j) a += P[tid * H_ + j] * hl[j];
    sc[tid] = a;
  }
  __syncthreads();
  if (tid == 0) {
    float m = sc[0];
    for (int t = 1; t < T_; ++t) m = fmaxf(m, sc[t]);
    float sum = 0.0f;
    for (int t = 0; t < T_; ++t) { aw[t] = expf(sc[t] - m); sum += aw[t]; }
    float inv = 1.0f / sum;
    for (int t = 0; t < T_; ++t) aw[t] *= inv;
  }
  __syncthreads();
  if (tid < H_) {
    float v = 0.0f;
    for (int t = 0; t < T_; ++t) v += aw[t] * O[t * H_ + tid];
    news[(size_t)blk * H_ + tid] = v;
  }
}

// =====================================================================================
// Kernel D: small per-stock GRU (price: in_dim=3, sequence: in_dim=64) + attention pool.
// 192 threads = one lane per gate; Whh row cached in VGPRs; state in LDS.
// =====================================================================================
__global__ void __launch_bounds__(192) rnn_kernel(const float* __restrict__ x, int in_dim,
                                                  const float* __restrict__ Wih,
                                                  const float* __restrict__ Whh,
                                                  const float* __restrict__ bih,
                                                  const float* __restrict__ bhh,
                                                  const float* __restrict__ Wa,
                                                  float* __restrict__ out_vec) {
  __shared__ __align__(16) float xb[D_ * 64];
  __shared__ float gi[D_ * G_];
  __shared__ float hB[H_];
  __shared__ float gs[G_];
  __shared__ float O[D_ * H_];
  __shared__ float P[D_ * H_];
  __shared__ float sc[D_];
  __shared__ float aw[D_];
  int s = blockIdx.x, tid = threadIdx.x;
  int g = tid;  // blockDim == 192 == G_

  for (int idx = tid; idx < D_ * in_dim; idx += 192)   // async global->LDS copy
    async_ld_b32(xb + idx, x + (size_t)s * D_ * in_dim + idx);
  if (tid < H_) hB[tid] = 0.0f;
  wait_async0();
  __syncthreads();

  float wh[64];
#pragma unroll
  for (int k = 0; k < 64; ++k) wh[k] = Whh[((size_t)s * G_ + g) * H_ + k];
  float bi = bih[(size_t)s * G_ + g];
  float bh = bhh[(size_t)s * G_ + g];

  float acc[D_];
#pragma unroll
  for (int d = 0; d < D_; ++d) acc[d] = bi;
  for (int f = 0; f < in_dim; ++f) {
    float wv = Wih[((size_t)s * G_ + g) * in_dim + f];
#pragma unroll
    for (int d = 0; d < D_; ++d) acc[d] += xb[d * in_dim + f] * wv;
  }
#pragma unroll
  for (int d = 0; d < D_; ++d) gi[d * G_ + g] = acc[d];
  __syncthreads();

  for (int t = 0; t < D_; ++t) {
    float gh = bh;
#pragma unroll
    for (int k = 0; k < 64; ++k) gh += wh[k] * hB[k];
    gs[g] = gh;
    __syncthreads();
    if (tid < H_) {
      int j = tid;
      float r = sigmoidf_(gi[t * G_ + j] + gs[j]);
      float z = sigmoidf_(gi[t * G_ + 64 + j] + gs[64 + j]);
      float n = tanhf(gi[t * G_ + 128 + j] + r * gs[128 + j]);
      float h = (1.0f - z) * n + z * hB[j];
      hB[j] = h;
      O[t * H_ + j] = h;
    }
    __syncthreads();
  }
  // attention pool
  for (int idx = tid; idx < D_ * H_; idx += 192) {
    int t = idx >> 6, j = idx & 63;
    float a = 0.0f;
    for (int k = 0; k < H_; ++k) a += O[t * H_ + k] * Wa[(size_t)s * H_ * H_ + k * H_ + j];
    P[idx] = tanhf(a);
  }
  __syncthreads();
  if (tid < D_) {
    float a = 0.0f;
    for (int j = 0; j < H_; ++j) a += P[tid * H_ + j] * hB[j];
    sc[tid] = a;
  }
  __syncthreads();
  if (tid == 0) {
    float m = sc[0];
    for (int t = 1; t < D_; ++t) m = fmaxf(m, sc[t]);
    float sum = 0.0f;
    for (int t = 0; t < D_; ++t) { aw[t] = expf(sc[t] - m); sum += aw[t]; }
    float inv = 1.0f / sum;
    for (int t = 0; t < D_; ++t) aw[t] *= inv;
  }
  __syncthreads();
  if (tid < H_) {
    float v = 0.0f;
    for (int t = 0; t < D_; ++t) v += aw[t] * O[t * H_ + tid];
    out_vec[(size_t)s * H_ + tid] = v;
  }
}

// =====================================================================================
// Kernel E: bilinear fusion. Streams bil_B (105 MB) once; one wave per (s,o);
// prefetch 2 KB ahead on the stream.
// feature[s,o] = tanh( sum_ij t[i] * B[s,o,i,j] * p[j] + b[s,o] )
// =====================================================================================
__global__ void __launch_bounds__(256) bilinear_kernel(const float* __restrict__ tv,
                                                       const float* __restrict__ pv,
                                                       const float* __restrict__ B,
                                                       const float* __restrict__ bb,
                                                       float* __restrict__ feature) {
  __shared__ float tbuf[H_];
  __shared__ float pbuf[H_];
  int s = blockIdx.x >> 3;
  int tid = threadIdx.x, lane = tid & 31, wid = tid >> 5;
  int o = ((blockIdx.x & 7) << 3) + wid;
  if (tid < H_) tbuf[tid] = tv[s * H_ + tid];
  else if (tid < 2 * H_) pbuf[tid - H_] = pv[s * H_ + tid - H_];
  __syncthreads();
  const float* Bp = B + ((size_t)s * H_ + o) * H_ * H_;
  float a = 0.0f;
  for (int c = lane; c < H_ * 16; c += 32) {  // float4 per lane: 512B/wave/iter
    int idx = c << 2;
    int i = idx >> 6, j = idx & 63;
    float4 v = *(const float4*)(Bp + idx);
    if (idx + 512 < H_ * H_) __builtin_prefetch(Bp + idx + 512, 0, 0);
    float ti = tbuf[i];
    a += ti * (v.x * pbuf[j] + v.y * pbuf[j + 1] + v.z * pbuf[j + 2] + v.w * pbuf[j + 3]);
  }
#pragma unroll
  for (int off = 16; off > 0; off >>= 1) a += __shfl_xor(a, off, 32);
  if (lane == 0) feature[s * H_ + o] = tanhf(a + bb[s * H_ + o]);
}

// =====================================================================================
// Kernel F: GAT heads (one block per head). GEMM1 h=x@W (WMMA), masked row softmax
// (one wave per row), GEMM2 att@h (WMMA, K padded 100->128 with zeros), ELU.
// h kept transposed in LDS (Ht[n][k]) so GEMM1 D-frags store as one v8h each and
// GEMM2 B-frags load as two ds_load_b128.
// =====================================================================================
__global__ void __launch_bounds__(256) gat_head_kernel(const float* __restrict__ feat,
                                                       const float* __restrict__ gatW,
                                                       const float* __restrict__ gata,
                                                       const float* __restrict__ adj,
                                                       float* __restrict__ heads) {
  __shared__ __align__(16) char smem[46080];
  _Float16* Fh  = (_Float16*)smem;               // [112][64] phase1 (A, row-major)
  _Float16* Wt  = (_Float16*)(smem + 14336);     // [64][64]  phase1 (Bt: Wt[n][k]=W[k][n])
  _Float16* A16 = (_Float16*)smem;               // [112][128] phase3 (overlaps Fh/Wt)
  float*    f12 = (float*)(smem + 28672);        // f1[100], f2[100]
  _Float16* Ht  = (_Float16*)(smem + 29472);     // [64][128]: Ht[n][i] = h[i][n]
  int head = blockIdx.x;
  int tid = threadIdx.x, lane = tid & 31, wid = tid >> 5;

  for (int idx = tid; idx < 112 * 64; idx += 256) {
    int m = idx >> 6;
    Fh[idx] = (_Float16)((m < S_) ? feat[idx] : 0.0f);
  }
  for (int idx = tid; idx < 64 * 64; idx += 256) {
    int k = idx >> 6, n = idx & 63;            // coalesced global read, transposed LDS write
    Wt[n * 64 + k] = (_Float16)gatW[head * 64 * 64 + idx];
  }
  for (int idx = tid; idx < 64 * 128; idx += 256) Ht[idx] = (_Float16)0.0f;  // zero K padding
  __syncthreads();
  // GEMM1: h = Fh @ W  (7 Mtiles x 4 Ntiles), store transposed into Ht
  for (int tile = wid; tile < 28; tile += 8) {
    int mt = tile >> 2, nt = tile & 3;
    v8f acc = {};
#pragma unroll
    for (int ks = 0; ks < 2; ++ks) {
      v16h a = load_frag_A(Fh, 64, mt * 16, ks * 32, lane);
      v16h b = load_frag_Bt(Wt, 64, ks * 32, nt * 16, lane);
      acc = __builtin_amdgcn_wmma_f32_16x16x32_f16(false, a, false, b, (short)0, acc, false,
                                                   false);
    }
    int nl = lane & 15, mb = (lane >> 4) << 3;
    v8h hv;
#pragma unroll
    for (int r = 0; r < 8; ++r) hv[r] = (_Float16)acc[r];
    *(v8h*)(Ht + (nt * 16 + nl) * 128 + mt * 16 + mb) = hv;   // contiguous 16B store
  }
  __syncthreads();
  const float* av = gata + head * 2 * H_;
  for (int i = tid; i < S_; i += 256) {
    float s1 = 0.0f, s2 = 0.0f;
    for (int k = 0; k < 64; ++k) {
      float hv = (float)Ht[k * 128 + i];   // h[i][k]
      s1 += hv * av[k];
      s2 += hv * av[64 + k];
    }
    f12[i] = s1;
    f12[100 + i] = s2;
  }
  __syncthreads();
  for (int idx = tid; idx < 112 * 128; idx += 256) A16[idx] = (_Float16)0.0f;
  __syncthreads();
  // masked attention softmax, one wave per row
  for (int i = wid; i < S_; i += 8) {
    float fi = f12[i];
    float v[4];
#pragma unroll
    for (int q = 0; q < 4; ++q) {
      int j = lane + 32 * q;
      if (j < S_) {
        float e = fi + f12[100 + j];
        e = (e > 0.0f) ? e : ALPHA_ * e;
        v[q] = (adj[i * S_ + j] > 0.0f) ? e : -9e15f;
      } else v[q] = -3.0e38f;
    }
    float m = fmaxf(fmaxf(v[0], v[1]), fmaxf(v[2], v[3]));
#pragma unroll
    for (int off = 16; off > 0; off >>= 1) m = fmaxf(m, __shfl_xor(m, off, 32));
    float sum = 0.0f;
#pragma unroll
    for (int q = 0; q < 4; ++q) { v[q] = expf(v[q] - m); sum += v[q]; }
#pragma unroll
    for (int off = 16; off > 0; off >>= 1) sum += __shfl_xor(sum, off, 32);
    float inv = 1.0f / sum;
#pragma unroll
    for (int q = 0; q < 4; ++q) {
      int j = lane + 32 * q;
      if (j < S_) A16[i * 128 + j] = (_Float16)(v[q] * inv);
    }
  }
  __syncthreads();
  // GEMM2: out = A16[112x128] @ h[128x64] (B from Ht), ELU, store head
  for (int tile = wid; tile < 28; tile += 8) {
    int mt = tile >> 2, nt = tile & 3;
    v8f acc = {};
#pragma unroll
    for (int ks = 0; ks < 4; ++ks) {
      v16h a = load_frag_A(A16, 128, mt * 16, ks * 32, lane);
      v16h b = load_frag_Bt(Ht, 128, ks * 32, nt * 16, lane);
      acc = __builtin_amdgcn_wmma_f32_16x16x32_f16(false, a, false, b, (short)0, acc, false,
                                                   false);
    }
    int nl = lane & 15, mb = (lane >> 4) << 3;
#pragma unroll
    for (int r = 0; r < 8; ++r) {
      int m = mt * 16 + mb + r;
      if (m < S_) {
        float x = acc[r];
        heads[(size_t)head * S_ * H_ + m * H_ + nt * 16 + nl] =
            (x > 0.0f) ? x : (expf(x) - 1.0f);
      }
    }
  }
}

// =====================================================================================
// Kernel G: output GAT (N=2), blend linear, double softmax + CE loss. Single block.
// d_out = [loss, output(100x2)]
// =====================================================================================
__global__ void __launch_bounds__(256) final_kernel(const float* __restrict__ heads,
                                                    const float* __restrict__ feat,
                                                    const float* __restrict__ outW,
                                                    const float* __restrict__ outa,
                                                    const float* __restrict__ blW,
                                                    const float* __restrict__ blb,
                                                    const float* __restrict__ adj,
                                                    const int* __restrict__ label,
                                                    float* __restrict__ out) {
  __shared__ float g[S_ * 2];
  __shared__ float f12[2 * S_];
  __shared__ float E[S_ * S_];
  __shared__ float logits[S_ * 2];
  int tid = threadIdx.x, lane = tid & 31, wid = tid >> 5;

  for (int idx = tid; idx < 2 * S_; idx += 256) {   // g = concat(heads) @ out_W
    int i = idx >> 1, c = idx & 1;
    float a = 0.0f;
    for (int k = 0; k < NH_ * H_; ++k) {
      int hd = k >> 6, kk = k & 63;
      a += heads[(size_t)hd * S_ * H_ + i * H_ + kk] * outW[k * 2 + c];
    }
    g[idx] = a;
  }
  __syncthreads();
  for (int i = tid; i < S_; i += 256) {
    f12[i] = g[i * 2] * outa[0] + g[i * 2 + 1] * outa[1];
    f12[S_ + i] = g[i * 2] * outa[2] + g[i * 2 + 1] * outa[3];
  }
  __syncthreads();
  for (int i = wid; i < S_; i += 8) {
    float fi = f12[i];
    float v[4];
#pragma unroll
    for (int q = 0; q < 4; ++q) {
      int j = lane + 32 * q;
      if (j < S_) {
        float e = fi + f12[S_ + j];
        e = (e > 0.0f) ? e : ALPHA_ * e;
        v[q] = (adj[i * S_ + j] > 0.0f) ? e : -9e15f;
      } else v[q] = -3.0e38f;
    }
    float m = fmaxf(fmaxf(v[0], v[1]), fmaxf(v[2], v[3]));
#pragma unroll
    for (int off = 16; off > 0; off >>= 1) m = fmaxf(m, __shfl_xor(m, off, 32));
    float sum = 0.0f;
#pragma unroll
    for (int q = 0; q < 4; ++q) { v[q] = expf(v[q] - m); sum += v[q]; }
#pragma unroll
    for (int off = 16; off > 0; off >>= 1) sum += __shfl_xor(sum, off, 32);
    float inv = 1.0f / sum;
#pragma unroll
    for (int q = 0; q < 4; ++q) {
      int j = lane + 32 * q;
      if (j < S_) E[i * S_ + j] = v[q] * inv;
    }
  }
  __syncthreads();
  for (int idx = tid; idx < 2 * S_; idx += 256) {
    int i = idx >> 1, c = idx & 1;
    float a = 0.0f;
    for (int j = 0; j < S_; ++j) a += E[i * S_ + j] * g[j * 2 + c];
    float x2 = (a > 0.0f) ? a : (expf(a) - 1.0f);       // elu
    float o1 = blb[c];
    for (int k = 0; k < H_; ++k) o1 += feat[i * H_ + k] * blW[k * 2 + c];
    logits[idx] = x2 + tanhf(o1);
  }
  __syncthreads();
  for (int i = tid; i < S_; i += 256) {
    float l0 = logits[i * 2], l1 = logits[i * 2 + 1];
    float m = fmaxf(l0, l1);
    float e0 = expf(l0 - m), e1 = expf(l1 - m);
    float inv = 1.0f / (e0 + e1);
    float p0 = e0 * inv, p1 = e1 * inv;
    out[1 + i * 2] = p0;
    out[1 + i * 2 + 1] = p1;
    logits[i * 2] = p0;       // reuse for loss (reference double-softmax bug kept)
    logits[i * 2 + 1] = p1;
  }
  __syncthreads();
  if (tid == 0) {
    float loss = 0.0f;
    for (int i = 0; i < S_; ++i) {
      float p0 = logits[i * 2], p1 = logits[i * 2 + 1];
      float m = fmaxf(p0, p1);
      float lse = m + logf(expf(p0 - m) + expf(p1 - m));
      float sel = (label[i] == 0) ? p0 : p1;
      loss += sel - lse;
    }
    out[0] = -loss / (float)S_;
  }
}

// =====================================================================================
extern "C" void kernel_launch(void* const* d_in, const int* in_sizes, int n_in, void* d_out,
                              int out_size, void* d_ws, size_t ws_size, hipStream_t stream) {
  const float* text_input  = (const float*)d_in[0];
  const float* price_input = (const float*)d_in[1];
  const int*   label       = (const int*)d_in[2];
  const float* adj         = (const float*)d_in[3];
  // d_in[4] = train (eval path: dropout identity)
  const float* pg_Wih = (const float*)d_in[5];
  const float* pg_Whh = (const float*)d_in[6];
  const float* pg_bih = (const float*)d_in[7];
  const float* pg_bhh = (const float*)d_in[8];
  const float* pa_W   = (const float*)d_in[9];
  const float* tg_Wih = (const float*)d_in[10];
  const float* tg_Whh = (const float*)d_in[11];
  const float* tg_bih = (const float*)d_in[12];
  const float* tg_bhh = (const float*)d_in[13];
  const float* ta_W   = (const float*)d_in[14];
  const float* sg_Wih = (const float*)d_in[15];
  const float* sg_Whh = (const float*)d_in[16];
  const float* sg_bih = (const float*)d_in[17];
  const float* sg_bhh = (const float*)d_in[18];
  const float* sa_W   = (const float*)d_in[19];
  const float* bil_B  = (const float*)d_in[20];
  const float* bil_b  = (const float*)d_in[21];
  const float* bl_W   = (const float*)d_in[22];
  const float* bl_b   = (const float*)d_in[23];
  const float* gat_W  = (const float*)d_in[24];
  const float* gat_a  = (const float*)d_in[25];
  const float* out_W  = (const float*)d_in[26];
  const float* out_a  = (const float*)d_in[27];

  // workspace layout (floats), ~63 MB total
  float* ws        = (float*)d_ws;
  float* gi_text   = ws;                                    // S*D*T*G = 11,520,000
  float* outs_text = gi_text + (size_t)S_ * D_ * T_ * G_;   // S*D*T*H =  3,840,000
  float* hlast_t   = outs_text + (size_t)S_ * D_ * T_ * H_; // S*D*H   =    128,000
  float* news      = hlast_t + (size_t)S_ * D_ * H_;        // S*D*H   =    128,000
  float* price_vec = news + (size_t)S_ * D_ * H_;           // S*H
  float* text_vec  = price_vec + S_ * H_;
  float* feature   = text_vec + S_ * H_;
  float* heads     = feature + S_ * H_;                     // NH*S*H = 51,200

  text_gi_kernel<<<S_ * D_, 256, 0, stream>>>(text_input, tg_Wih, tg_bih, gi_text);
  text_rec_kernel<<<S_, 256, 0, stream>>>(gi_text, tg_Whh, tg_bhh, outs_text, hlast_t);
  attn_text_kernel<<<S_ * D_, 128, 0, stream>>>(outs_text, hlast_t, ta_W, news);
  rnn_kernel<<<S_, 192, 0, stream>>>(price_input, 3, pg_Wih, pg_Whh, pg_bih, pg_bhh, pa_W,
                                     price_vec);
  rnn_kernel<<<S_, 192, 0, stream>>>(news, H_, sg_Wih, sg_Whh, sg_bih, sg_bhh, sa_W, text_vec);
  bilinear_kernel<<<(S_ * H_) / 8, 256, 0, stream>>>(text_vec, price_vec, bil_B, bil_b, feature);
  gat_head_kernel<<<NH_, 256, 0, stream>>>(feature, gat_W, gat_a, adj, heads);
  final_kernel<<<1, 256, 0, stream>>>(heads, feature, out_W, out_a, bl_W, bl_b, adj, label,
                                      (float*)d_out);
  (void)in_sizes; (void)n_in; (void)out_size; (void)ws_size;
}
